// LinearAttention1_4200478015532
// MI455X (gfx1250) — matrix operand order repaired and checked by reading, not compiled, not run on hardware
//
#include <hip/hip_runtime.h>
#include <hip/hip_bf16.h>
#include <cstdint>

// ---------------- problem constants ----------------
constexpr int BATCH = 16;
constexpr int C     = 384;
constexpr int HIMG  = 56;
constexpr int WIMG  = 56;
constexpr int NPIX  = HIMG * WIMG;     // 3136
constexpr int NH    = 6;
constexpr int HD    = 64;              // C / NH
constexpr int CH    = C / 2;           // 192 input half-channels per group
constexpr float BN_EPS = 1e-5f;

// ---------------- WMMA types ----------------
typedef __attribute__((ext_vector_type(16))) __bf16 v16bf;
typedef __attribute__((ext_vector_type(8)))  __bf16 v8bf;
typedef __attribute__((ext_vector_type(8)))  float  v8f;

__device__ __forceinline__ v8f wmma_bf16(const v16bf& a, const v16bf& b, const v8f& c) {
    return __builtin_amdgcn_wmma_f32_16x16x32_bf16(false, a, false, b, (short)0, c, false, false);
}

// load a 16-bf16 A/B fragment from two contiguous 16-byte chunks (gap16 apart for A)
__device__ __forceinline__ v16bf frag_from2(const __bf16* p0, const __bf16* p1) {
    union { v16bf v; v8bf h[2]; } u;
    u.h[0] = *(const v8bf*)p0;
    u.h[1] = *(const v8bf*)p1;
    return u.v;
}

// LDS byte offset of a __shared__ object (AS3 pointer value == LDS offset)
__device__ __forceinline__ uint32_t lds_byte_off(const void* p) {
    typedef __attribute__((address_space(3))) const void lds_cv_t;
    return (uint32_t)(uintptr_t)(lds_cv_t*)p;
}

// async copy of 16 bytes global -> LDS (CDNA5 GLOBAL_LOAD_ASYNC_TO_LDS_B128)
__device__ __forceinline__ void async_g2l_b128(uint32_t lds_off, const void* gptr) {
    uint64_t ga = (uint64_t)(uintptr_t)gptr;
    asm volatile("global_load_async_to_lds_b128 %0, %1, off"
                 :: "v"(lds_off), "v"(ga) : "memory");
}
__device__ __forceinline__ void wait_async0() {
    asm volatile("s_wait_asynccnt 0" ::: "memory");
}

// =====================================================================
// Kernel 0a: BN coefficient precompute: coef[2*ch] = scale, coef[2*ch+1] = shift
// =====================================================================
__global__ __launch_bounds__(256) void bncoef_kernel(
    const float* __restrict__ gamma, const float* __restrict__ beta,
    const float* __restrict__ mean, const float* __restrict__ var,
    float* __restrict__ coef, int nch)
{
    int i = blockIdx.x * 256 + threadIdx.x;
    if (i >= nch) return;
    float sc = gamma[i] * rsqrtf(var[i] + BN_EPS);
    coef[2 * i]     = sc;
    coef[2 * i + 1] = beta[i] - mean[i] * sc;
}

// =====================================================================
// Kernel 0b: convert qk weights to bf16
// =====================================================================
__global__ __launch_bounds__(256) void wconv_kernel(
    const float* __restrict__ w, __bf16* __restrict__ wb, int n)
{
    int i = blockIdx.x * 256 + threadIdx.x;
    if (i < n) wb[i] = (__bf16)w[i];
}

// =====================================================================
// Kernel 0c: x -> x_t[b][g][n][192] bf16 (transposed) AND x_bf16[b][c][n]
// =====================================================================
__global__ __launch_bounds__(256) void xt_kernel(
    const float* __restrict__ x, __bf16* __restrict__ x_t,
    __bf16* __restrict__ x_b)
{
    __shared__ __bf16 t[32][33];
    const int tx = threadIdx.x;            // 0..31
    const int ty = threadIdx.y;            // 0..7
    const int ntile = blockIdx.x;          // 0..97
    const int itile = blockIdx.y;          // 0..5
    const int bg = blockIdx.z;             // 0..31
    const int b = bg >> 1, g = bg & 1;

    const float* xp = x + ((size_t)b * C + (size_t)g * CH) * NPIX;    // [i][n]
    __bf16* xbp = x_b + ((size_t)b * C + (size_t)g * CH) * NPIX;      // [i][n] bf16
    #pragma unroll
    for (int r = 0; r < 4; ++r) {
        const int i = itile * 32 + ty + r * 8;
        const int n = ntile * 32 + tx;
        const __bf16 v = (__bf16)xp[(size_t)i * NPIX + n];
        t[ty + r * 8][tx] = v;
        xbp[(size_t)i * NPIX + n] = v;     // straight bf16 copy (coalesced in n)
    }
    __syncthreads();
    __bf16* op = x_t + (size_t)bg * NPIX * CH;                        // [n][192]
    #pragma unroll
    for (int r = 0; r < 4; ++r) {
        const int nl = ty + r * 8;
        op[(size_t)(ntile * 32 + nl) * CH + itile * 32 + tx] = t[tx][nl];
    }
}

// =====================================================================
// Kernel 1: depthwise 3x3 conv + BN -> writes d_out (pe term)
// =====================================================================
__global__ __launch_bounds__(256) void pe_kernel(
    const float* __restrict__ x, const float* __restrict__ w,
    const float* __restrict__ gamma, const float* __restrict__ beta,
    const float* __restrict__ mean, const float* __restrict__ var,
    float* __restrict__ out)
{
    size_t idx = (size_t)blockIdx.x * 256 + threadIdx.x;
    const size_t total = (size_t)BATCH * C * NPIX;
    if (idx >= total) return;
    int xx = (int)(idx % WIMG);
    size_t t = idx / WIMG;
    int yy = (int)(t % HIMG); t /= HIMG;
    int ch = (int)(t % C);
    int b  = (int)(t / C);

    const float* xp = x + ((size_t)b * C + ch) * NPIX;
    const float* wp = w + (size_t)ch * 9;
    float acc = 0.f;
    #pragma unroll
    for (int dy = 0; dy < 3; ++dy) {
        int y = yy + dy - 1;
        if (y < 0 || y >= HIMG) continue;
        #pragma unroll
        for (int dx = 0; dx < 3; ++dx) {
            int x2 = xx + dx - 1;
            if (x2 < 0 || x2 >= WIMG) continue;
            acc += xp[y * WIMG + x2] * wp[dy * 3 + dx];
        }
    }
    float sc = gamma[ch] * rsqrtf(var[ch] + BN_EPS);
    out[idx] = acc * sc + (beta[ch] - mean[ch] * sc);
}

// =====================================================================
// Kernel 2: QK projection GEMM (WMMA bf16, async-to-LDS staged B tiles)
//   4 waves/block; block tile: 64 out-channels x 64 pixels; K = 192
//   B tile (64 n x 32 i bf16 = 4KB) double-buffered in LDS via
//   global_load_async_to_lds_b128.
// =====================================================================
__global__ __launch_bounds__(128) void qk_gemm_kernel(
    const __bf16* __restrict__ x_t, const __bf16* __restrict__ qk_wb,
    const float* __restrict__ coef,
    __bf16* __restrict__ q_t, __bf16* __restrict__ k_out)
{
    __shared__ __bf16 lds[2][64 * 32];

    const int tid   = threadIdx.x;        // 0..127
    const int lane  = tid & 31;
    const int wv    = tid >> 5;           // 0..3 -> M sub-tile
    const int nt    = blockIdx.x;         // 0..48  -> n base = nt*64
    const int mt64  = blockIdx.y;         // 0..5   -> o base = mt64*64
    const int bg    = blockIdx.z;         // 0..31
    const int b     = bg >> 1;
    const int g     = bg & 1;
    const int row_l = lane & 15;
    const int half  = lane >> 4;
    const int nbase = nt * 64;
    const int obase = mt64 * 64 + wv * 16;

    const __bf16* xt_bg = x_t + (size_t)bg * NPIX * CH;      // [n][192]
    const __bf16* wg    = qk_wb + (size_t)g * C * CH;        // [o][i] bf16

    // ---- async stage one 64n x 32i tile into LDS buffer `buf` ----
    auto issue_tile = [&](int kt, int buf) {
        #pragma unroll
        for (int i = 0; i < 2; ++i) {
            const int seg  = tid + i * 128;   // 0..255 -> 16B segments
            const int rowN = seg >> 2;        // 0..63
            const int part = seg & 3;         // 0..3 (8 bf16 each)
            const __bf16* gp = xt_bg + (size_t)(nbase + rowN) * CH + kt * 32 + part * 8;
            const uint32_t lo = lds_byte_off(&lds[buf][rowN * 32 + part * 8]);
            async_g2l_b128(lo, gp);
        }
    };

    issue_tile(0, 0);

    v8f acc[4] = {};
    for (int kt = 0; kt < CH / 32; ++kt) {
        // A fragment: bf16 weights, row o fixed per lane, contiguous i
        const __bf16* wrow = wg + (size_t)(obase + row_l) * CH + kt * 32 + half * 8;
        const v16bf a = frag_from2(wrow, wrow + 16);

        wait_async0();        // my async ops for tile kt are done
        __syncthreads();      // whole tile visible; all waves done with other buffer
        if (kt + 1 < CH / 32) issue_tile(kt + 1, (kt + 1) & 1);

        const __bf16* lbuf = lds[kt & 1];
        #pragma unroll
        for (int sub = 0; sub < 4; ++sub) {
            const __bf16* lp = lbuf + (sub * 16 + row_l) * 32 + half * 8;
            acc[sub] = wmma_bf16(a, frag_from2(lp, lp + 16), acc[sub]);
        }
    }

    // epilogue: BN + elu + 1, store bf16
    #pragma unroll
    for (int sub = 0; sub < 4; ++sub) {
        const int col = nbase + sub * 16 + row_l;    // pixel n
        __bf16 pk[8];
        #pragma unroll
        for (int r = 0; r < 8; ++r) {
            const int row = obase + half * 8 + r;    // channel in [0,384)
            const int ch2 = g * C + row;             // BN channel
            float v = acc[sub][r] * coef[2 * ch2] + coef[2 * ch2 + 1];
            v = (v > 0.f) ? (v + 1.f) : __expf(v);
            pk[r] = (__bf16)v;
        }
        if (g == 0) {
            const int row0 = obase + half * 8;       // 8 consecutive channels
            const int h = row0 >> 6, d0 = row0 & 63; // stays inside one head
            v8bf* dst = (v8bf*)&q_t[(((size_t)b * NH + h) * NPIX + col) * HD + d0];
            *dst = *(v8bf*)pk;
        } else {
            #pragma unroll
            for (int r = 0; r < 8; ++r) {
                const int row = obase + half * 8 + r;
                k_out[((size_t)b * C + row) * NPIX + col] = pk[r];
            }
        }
    }
}

// =====================================================================
// Kernel 3: kmean[b*C + ch] = mean over n of k
// =====================================================================
__global__ __launch_bounds__(128) void kmean_kernel(
    const __bf16* __restrict__ k_ws, float* __restrict__ kmean)
{
    __shared__ float red[128];
    const int row = blockIdx.x;                    // 0 .. B*C-1
    const __bf16* kr = k_ws + (size_t)row * NPIX;
    float s = 0.f;
    for (int i = threadIdx.x; i < NPIX; i += 128) s += (float)kr[i];
    red[threadIdx.x] = s;
    __syncthreads();
    for (int off = 64; off > 0; off >>= 1) {
        if ((int)threadIdx.x < off) red[threadIdx.x] += red[threadIdx.x + off];
        __syncthreads();
    }
    if (threadIdx.x == 0) kmean[row] = red[0] / (float)NPIX;
}

// =====================================================================
// Kernel 4: kvT[b,h,e,d] = (1/N) * sum_n k[d,n] * v[e,n]   (WMMA bf16)
//   one block per (b,h); 4 waves; wave w computes d-tile w (16x64), K = 3136
//   both operands bf16, fully contiguous fragment loads; output stored
//   TRANSPOSED (e-major) in bf16 for contiguous attn B-frags.
// =====================================================================
__global__ __launch_bounds__(128) void kv_kernel(
    const __bf16* __restrict__ k_ws, const __bf16* __restrict__ x_b,
    __bf16* __restrict__ kvT)
{
    const int lane  = threadIdx.x & 31;
    const int wv    = threadIdx.x >> 5;    // 0..3 -> d tile
    const int bh    = blockIdx.x;          // 0..95
    const int b     = bh / NH, h = bh % NH;
    const int row_l = lane & 15;
    const int half  = lane >> 4;

    const __bf16* kk = k_ws + ((size_t)b * C + (size_t)h * HD) * NPIX; // [d][n]
    const __bf16* vv = x_b  + ((size_t)b * C + (size_t)h * HD) * NPIX; // [e][n]

    v8f acc[4] = {};
    for (int kt = 0; kt < NPIX / 32; ++kt) {
        const __bf16* krow = kk + (size_t)(wv * 16 + row_l) * NPIX + kt * 32 + half * 8;
        const v16bf a = frag_from2(krow, krow + 16);
        #pragma unroll
        for (int sub = 0; sub < 4; ++sub) {
            const __bf16* vrow = vv + (size_t)(sub * 16 + row_l) * NPIX + kt * 32 + half * 8;
            acc[sub] = wmma_bf16(a, frag_from2(vrow, vrow + 16), acc[sub]);
        }
    }
    const float s2 = 1.0f / (float)NPIX;
    #pragma unroll
    for (int sub = 0; sub < 4; ++sub) {
        const int e  = sub * 16 + row_l;
        const int d0 = wv * 16 + half * 8;
        __bf16 pk[8];
        #pragma unroll
        for (int r = 0; r < 8; ++r) pk[r] = (__bf16)(acc[sub][r] * s2);
        *(v8bf*)&kvT[((size_t)bh * HD + e) * HD + d0] = *(v8bf*)pk;
    }
}

// =====================================================================
// Kernel 5: den[b,h,n] = sum_d q_t[b,h,n,d] * kmean[b, h*64+d] + 1e-6
// =====================================================================
__global__ __launch_bounds__(256) void den_kernel(
    const __bf16* __restrict__ q_t, const float* __restrict__ kmean,
    float* __restrict__ den)
{
    size_t idx = (size_t)blockIdx.x * 256 + threadIdx.x;
    const size_t total = (size_t)BATCH * NH * NPIX;
    if (idx >= total) return;
    const int n  = (int)(idx % NPIX);
    const int bh = (int)(idx / NPIX);
    const int b  = bh / NH, h = bh % NH;
    const __bf16* qr = q_t + ((size_t)bh * NPIX + n) * HD;
    const float*  km = kmean + (size_t)b * C + (size_t)h * HD;
    float s = 0.f;
    #pragma unroll 8
    for (int d = 0; d < HD; ++d) s += (float)qr[d] * km[d];
    den[idx] = s + 1e-6f;
}

// =====================================================================
// Kernel 6: num[n,e] = sum_d q_t[n,d] * kvT[e,d]; out += num/den (WMMA bf16)
// =====================================================================
__global__ __launch_bounds__(128) void attn_kernel(
    const __bf16* __restrict__ q_t, const __bf16* __restrict__ kvT,
    const float* __restrict__ den, float* __restrict__ out)
{
    const int lane  = threadIdx.x & 31;
    const int wv    = threadIdx.x >> 5;
    const int mt    = blockIdx.x * 4 + wv;   // 0..195  n-tile
    const int bh    = blockIdx.y;            // 0..95
    const int b     = bh / NH, h = bh % NH;
    const int row_l = lane & 15;
    const int half  = lane >> 4;

    const __bf16* qq  = q_t + (size_t)bh * NPIX * HD;  // [n][d]
    const __bf16* kvp = kvT + (size_t)bh * HD * HD;    // [e][d]

    v8f acc[4] = {};
    #pragma unroll
    for (int kt = 0; kt < 2; ++kt) {
        const __bf16* qrow = qq + (size_t)(mt * 16 + row_l) * HD + kt * 32 + half * 8;
        const v16bf a = frag_from2(qrow, qrow + 16);
        #pragma unroll
        for (int sub = 0; sub < 4; ++sub) {
            const __bf16* lp = kvp + (size_t)(sub * 16 + row_l) * HD + kt * 32 + half * 8;
            acc[sub] = wmma_bf16(a, frag_from2(lp, lp + 16), acc[sub]);
        }
    }
    #pragma unroll
    for (int sub = 0; sub < 4; ++sub) {
        #pragma unroll
        for (int r = 0; r < 8; ++r) {
            const int m = mt * 16 + half * 8 + r;  // pixel n
            const int e = sub * 16 + row_l;        // dim within head
            const float dv = den[(size_t)bh * NPIX + m];
            const size_t oidx = ((size_t)b * C + (size_t)h * HD + e) * NPIX + m;
            out[oidx] += acc[sub][r] / dv;
        }
    }
}

// =====================================================================
// host launcher
// =====================================================================
extern "C" void kernel_launch(void* const* d_in, const int* in_sizes, int n_in,
                              void* d_out, int out_size, void* d_ws, size_t ws_size,
                              hipStream_t stream) {
    const float* x        = (const float*)d_in[0];
    const float* qk_w     = (const float*)d_in[1];
    const float* qk_gamma = (const float*)d_in[2];
    const float* qk_beta  = (const float*)d_in[3];
    const float* qk_mean  = (const float*)d_in[4];
    const float* qk_var   = (const float*)d_in[5];
    const float* pe_w     = (const float*)d_in[6];
    const float* pe_gamma = (const float*)d_in[7];
    const float* pe_beta  = (const float*)d_in[8];
    const float* pe_mean  = (const float*)d_in[9];
    const float* pe_var   = (const float*)d_in[10];
    float* out = (float*)d_out;

    // workspace partition
    const size_t QT_BYTES = (size_t)BATCH * NH * NPIX * HD * sizeof(__bf16); // 38.5 MB
    const size_t K_BYTES  = (size_t)BATCH * C * NPIX * sizeof(__bf16);       // 38.5 MB
    const size_t XT_BYTES = (size_t)BATCH * 2 * NPIX * CH * sizeof(__bf16);  // 38.5 MB
    const size_t XB_BYTES = (size_t)BATCH * C * NPIX * sizeof(__bf16);       // 38.5 MB
    const size_t KM_BYTES = (size_t)BATCH * C * sizeof(float);
    const size_t KV_BYTES = (size_t)BATCH * NH * HD * HD * sizeof(__bf16);
    const size_t CO_BYTES = (size_t)2 * 2 * C * sizeof(float);               // 768*2 f32
    const size_t WB_BYTES = (size_t)2 * C * CH * sizeof(__bf16);             // 294 KB
    char* wsp = (char*)d_ws;
    __bf16* q_t   = (__bf16*)wsp;             wsp += QT_BYTES;
    __bf16* k_ws  = (__bf16*)wsp;             wsp += K_BYTES;
    __bf16* x_t   = (__bf16*)wsp;             wsp += XT_BYTES;
    __bf16* x_b   = (__bf16*)wsp;             wsp += XB_BYTES;
    float*  kmean = (float*)wsp;              wsp += KM_BYTES;
    __bf16* kvT   = (__bf16*)wsp;             wsp += KV_BYTES;
    float*  coef  = (float*)wsp;              wsp += CO_BYTES;
    __bf16* qk_wb = (__bf16*)wsp;             wsp += WB_BYTES;
    float*  den   = (float*)wsp;

    const size_t total_out = (size_t)BATCH * C * NPIX;

    // 0a) BN coefficients for the qk path (768 channels)
    bncoef_kernel<<<dim3(3), 256, 0, stream>>>(qk_gamma, qk_beta, qk_mean, qk_var,
                                               coef, 2 * C);
    // 0b) qk weights -> bf16
    wconv_kernel<<<dim3((2 * C * CH + 255) / 256), 256, 0, stream>>>(
        qk_w, qk_wb, 2 * C * CH);
    // 0c) x -> x_t (transposed bf16) and x_b (straight bf16)
    xt_kernel<<<dim3(NPIX / 32, CH / 32, BATCH * 2), dim3(32, 8), 0, stream>>>(
        x, x_t, x_b);

    // 1) pe term -> d_out
    pe_kernel<<<dim3((unsigned)((total_out + 255) / 256)), 256, 0, stream>>>(
        x, pe_w, pe_gamma, pe_beta, pe_mean, pe_var, out);

    // 2) qk projection GEMM (WMMA + async-to-LDS) -> q_t, k_ws
    qk_gemm_kernel<<<dim3(NPIX / 64, C / 64, BATCH * 2), 128, 0, stream>>>(
        x_t, qk_wb, coef, q_t, k_ws);

    // 3) k mean over pixels
    kmean_kernel<<<dim3(BATCH * C), 128, 0, stream>>>(k_ws, kmean);

    // 4) kv GEMM (WMMA, all-bf16 operands) -> kvT bf16
    kv_kernel<<<dim3(BATCH * NH), 128, 0, stream>>>(k_ws, x_b, kvT);

    // 5) denominator
    den_kernel<<<dim3((unsigned)(((size_t)BATCH * NH * NPIX + 255) / 256)), 256, 0, stream>>>(
        q_t, kmean, den);

    // 6) numerator GEMM (WMMA) + divide + accumulate into d_out
    attn_kernel<<<dim3(NPIX / 64, BATCH * NH), 128, 0, stream>>>(q_t, kvT, den, out);
}
